// LSTM_37443524887262
// MI455X (gfx1250) — compile-verified, matrix operand order checked by reading, not run on hardware
//
#include <hip/hip_runtime.h>
#include <hip/hip_bf16.h>

// ---------------------------------------------------------------------------
// Fused LSTM encoder/decoder for MI455X (gfx1250), wave32 + WMMA bf16.
// Grid: 313 blocks x 512 threads (16 waves); each block owns 64 agents for
// all 18 timesteps. Gates GEMM and output head both run on v_wmma.
// ---------------------------------------------------------------------------

#define T_STEPS   18
#define ENC_STEPS 8
#define NAG       20000
#define EMB       64
#define HID       128
#define G4        512      // 4*HID gate width
#define KTOT      192      // EMB + HID
#define MB        64       // agents per block
#define NTHREADS  512      // 16 waves

typedef __attribute__((ext_vector_type(16))) __bf16 v16bf;
typedef __attribute__((ext_vector_type(8)))  float  v8f;

// ---- dynamic LDS layout (bytes) ----
#define BW_OFF     0
#define BW_BYTES   (6 * 32 * 1024)             // 192 B-fragments, 1KB each = 196608
#define A_OFF      (BW_OFF + BW_BYTES)         // 196608 (16B aligned)
#define A_BYTES    (MB * KTOT * 2)             // 24576
#define BIAS_OFF   (A_OFF + A_BYTES)           // 221184
#define WEMB_OFF   (BIAS_OFF + G4 * 4)         // 223232
#define BEMB_OFF   (WEMB_OFF + EMB * 2 * 4)    // 223744
#define WOUTF_OFF  (BEMB_OFF + EMB * 4)        // 224000 (w_out B-frags, 4KB)
#define BOUT_OFF   (WOUTF_OFF + 4 * 1024)      // 228096
#define VELX_OFF   (BOUT_OFF + 32)             // 228128
#define VELY_OFF   (VELX_OFF + MB * 4)         // 228384
#define MASK_OFF   (VELY_OFF + MB * 4)         // 228640
#define SMEM_TOTAL (MASK_OFF + MB * 4)         // 228896 < 320KB

__device__ __forceinline__ unsigned short f2bf(float x) {
    unsigned u = __float_as_uint(x);
    unsigned r = (u + 0x7FFFu + ((u >> 16) & 1u)) >> 16;   // RNE
    return (unsigned short)r;
}
__device__ __forceinline__ float sigf(float x)  { return 1.0f / (1.0f + __expf(-x)); }
__device__ __forceinline__ float tanhft(float x){ return 2.0f * sigf(2.0f * x) - 1.0f; }

// Pack stacked [w_ih | w_hh]^T (192 x 512) as bf16 WMMA B-fragments in LDS,
// exact per-lane order of the 16-bit 32x16 B layout:
//   K(e,l) = 32*kb + (e<8 ? e : e+8) + 8*(l>=16),  N(l,nt) = 16*nt + (l&15)
// Each of the 512 threads owns one (l,e) slot; all index math hoisted.
__device__ void load_weights(char* smem,
                             const float* __restrict__ w_ih, const float* __restrict__ b_ih,
                             const float* __restrict__ w_hh, const float* __restrict__ b_hh,
                             int tid) {
    unsigned short* Bw = (unsigned short*)(smem + BW_OFF);
    float* biasC = (float*)(smem + BIAS_OFF);
    const int e  = tid & 15;
    const int l  = (tid >> 4) & 31;
    const int Kb = (e < 8 ? e : e + 8) + ((l >= 16) ? 8 : 0);
    const int no = l & 15;
#pragma unroll 4
    for (int f = 0; f < 192; ++f) {             // f = kb*32 + nt
        int K = ((f >> 5) << 5) + Kb;
        int n = (f & 31) * 16 + no;
        float v = (K < EMB) ? w_ih[n * EMB + K] : w_hh[n * HID + (K - EMB)];
        Bw[f * 512 + l * 16 + e] = f2bf(v);
    }
    if (tid < G4) biasC[tid] = b_ih[tid] + b_hh[tid];
}

__global__ void __launch_bounds__(NTHREADS)
lstm_fused(const float* __restrict__ obs,
           const float* __restrict__ w_emb, const float* __restrict__ b_emb,
           const float* __restrict__ w_ih_e, const float* __restrict__ b_ih_e,
           const float* __restrict__ w_hh_e, const float* __restrict__ b_hh_e,
           const float* __restrict__ w_ih_d, const float* __restrict__ b_ih_d,
           const float* __restrict__ w_hh_d, const float* __restrict__ b_hh_d,
           const float* __restrict__ w_out, const float* __restrict__ b_out,
           float* __restrict__ out) {
    extern __shared__ char smem[];
    const int tid  = threadIdx.x;
    const int lane = tid & 31;
    const int wv   = tid >> 5;       // wave id 0..15
    const int mt   = wv >> 2;        // M tile 0..3 (agents 16*mt .. 16*mt+15)
    const int hq   = wv & 3;         // hidden quarter: hidden tiles hq*2, hq*2+1
    const int agbase = blockIdx.x * MB;

    unsigned short* A  = (unsigned short*)(smem + A_OFF);
    unsigned short* Wof = (unsigned short*)(smem + WOUTF_OFF);
    float* biasC  = (float*)(smem + BIAS_OFF);
    float* s_wemb = (float*)(smem + WEMB_OFF);
    float* s_bemb = (float*)(smem + BEMB_OFF);
    float* s_bout = (float*)(smem + BOUT_OFF);
    float* s_velx = (float*)(smem + VELX_OFF);
    float* s_vely = (float*)(smem + VELY_OFF);
    float* s_mask = (float*)(smem + MASK_OFF);

    // ---- one-time init ----
    for (int i = tid; i < EMB * 2; i += NTHREADS) s_wemb[i] = w_emb[i];
    for (int i = tid; i < EMB;     i += NTHREADS) s_bemb[i] = b_emb[i];
    for (int i = tid; i < 5;       i += NTHREADS) s_bout[i] = b_out[i];
    for (int i = tid; i < MB * HID; i += NTHREADS) {      // h(0) = 0 in A
        int a = i >> 7, k = i & 127;
        A[a * KTOT + EMB + k] = 0;
    }
    // pack w_out (5x128, zero-padded to 16 cols) as 4 B-fragments (kb = K/32)
    for (int idx = tid; idx < 4 * 512; idx += NTHREADS) {
        int e = idx & 15, l = (idx >> 4) & 31, kb = idx >> 9;
        int K = 32 * kb + (e < 8 ? e : e + 8) + ((l >= 16) ? 8 : 0);
        int n = l & 15;
        float v = (n < 5) ? w_out[n * HID + K] : 0.0f;
        Wof[kb * 512 + l * 16 + e] = f2bf(v);
    }
    load_weights(smem, w_ih_e, b_ih_e, w_hh_e, b_hh_e, tid);

    float c_reg[2][8], h_reg[2][8];
#pragma unroll
    for (int u = 0; u < 2; ++u)
#pragma unroll
        for (int r = 0; r < 8; ++r) { c_reg[u][r] = 0.0f; h_reg[u][r] = 0.0f; }

    const int rbase = (lane >= 16) ? 8 : 0;

    for (int t = 0; t < T_STEPS; ++t) {
        __syncthreads();                                   // prev-step readers done
        if (t == ENC_STEPS)
            load_weights(smem, w_ih_d, b_ih_d, w_hh_d, b_hh_d, tid);

        // ---- velocity + NaN mask ----
        if (tid < MB) {
            int ga = agbase + tid;
            float o1x = 0.f, o1y = 0.f, o2x = 0.f, o2y = 0.f;
            if (ga < NAG) {
                const float* p1 = obs + ((size_t)t * NAG + ga) * 2;
                const float* p2 = obs + ((size_t)(t + 1) * NAG + ga) * 2;
                o1x = p1[0]; o1y = p1[1]; o2x = p2[0]; o2y = p2[1];
            }
            float m = ((o1x == o1x) && (o2x == o2x)) ? 1.0f : 0.0f;
            s_mask[tid] = m;
            s_velx[tid] = (m != 0.0f) ? (o2x - o1x) : 0.0f;
            s_vely[tid] = (m != 0.0f) ? (o2y - o1y) : 0.0f;
        }
        __syncthreads();                                   // vel/mask (+weights) ready

        // ---- embedding: emb = relu(4*vel @ w_emb^T + b) -> A[:, 0:64] bf16 ----
        for (int i = tid; i < MB * EMB; i += NTHREADS) {
            int a = i >> 6, j = i & 63;
            float e = 4.0f * s_velx[a] * s_wemb[j * 2 + 0] +
                      4.0f * s_vely[a] * s_wemb[j * 2 + 1] + s_bemb[j];
            A[a * KTOT + j] = f2bf(e > 0.0f ? e : 0.0f);
        }
        __syncthreads();                                   // A ready for WMMA

        // ---- gates = A(64x192) @ W(192x512) + bias, via WMMA bf16 ----
        v8f acc[2][4];
#pragma unroll
        for (int u = 0; u < 2; ++u) {
            int ht = hq * 2 + u;
#pragma unroll
            for (int g = 0; g < 4; ++g) {
                float bv = biasC[g * HID + ht * 16 + (lane & 15)];
                v8f z;
#pragma unroll
                for (int r = 0; r < 8; ++r) z[r] = bv;
                acc[u][g] = z;
            }
        }
#pragma unroll
        for (int kb = 0; kb < 6; ++kb) {
            // A fragment: 2 contiguous 16B chunks per lane from row-major A
            int row = mt * 16 + (lane & 15);
            int cb  = 32 * kb + ((lane >= 16) ? 8 : 0);
            const char* ap = smem + A_OFF + (size_t)(row * KTOT + cb) * 2;
            union { v16bf v; uint4 q[2]; } af;
            af.q[0] = *(const uint4*)(ap);
            af.q[1] = *(const uint4*)(ap + 32);
#pragma unroll
            for (int u = 0; u < 2; ++u) {
                int ht = hq * 2 + u;
#pragma unroll
                for (int g = 0; g < 4; ++g) {
                    int nt = g * 8 + ht;
                    const char* bp = smem + BW_OFF + (size_t)((kb * 32 + nt) * 32 + lane) * 32;
                    union { v16bf v; uint4 q[2]; } bf_;
                    bf_.q[0] = *(const uint4*)(bp);
                    bf_.q[1] = *(const uint4*)(bp + 16);
                    acc[u][g] = __builtin_amdgcn_wmma_f32_16x16x32_bf16(
                        false, af.v, false, bf_.v, (short)0, acc[u][g], false, false);
                }
            }
        }

        // ---- elementwise LSTM cell (all 4 gates of these columns are local) ----
#pragma unroll
        for (int u = 0; u < 2; ++u) {
#pragma unroll
            for (int r = 0; r < 8; ++r) {
                float iv = sigf(acc[u][0][r]);
                float fv = sigf(acc[u][1][r]);
                float gv = tanhft(acc[u][2][r]);
                float ov = sigf(acc[u][3][r]);
                float c2 = fv * c_reg[u][r] + iv * gv;
                float h2 = ov * tanhft(c2);
                float m  = s_mask[mt * 16 + rbase + r];
                c_reg[u][r] = fmaf(m, c2 - c_reg[u][r], c_reg[u][r]);
                h_reg[u][r] = fmaf(m, h2 - h_reg[u][r], h_reg[u][r]);
            }
        }
        __syncthreads();                                   // all gate A/B reads done

        // ---- write blended h back into A[:, 64:192] ----
#pragma unroll
        for (int u = 0; u < 2; ++u) {
            int ht  = hq * 2 + u;
            int col = EMB + ht * 16 + (lane & 15);
#pragma unroll
            for (int r = 0; r < 8; ++r) {
                int row = mt * 16 + rbase + r;
                A[row * KTOT + col] = f2bf(h_reg[u][r]);
            }
        }
        __syncthreads();                                   // h visible

        // ---- output head via WMMA: n = h(16x128) @ w_out^T(128x16pad) ----
        if (hq == 0) {                                     // wave-uniform branch
            int j = lane & 15;
            float bj = s_bout[j < 5 ? j : 0];
            v8f hacc;
#pragma unroll
            for (int r = 0; r < 8; ++r) hacc[r] = bj;
#pragma unroll
            for (int kb = 2; kb < 6; ++kb) {               // h occupies K = 64..191
                int row = mt * 16 + (lane & 15);
                int cb  = 32 * kb + ((lane >= 16) ? 8 : 0);
                const char* ap = smem + A_OFF + (size_t)(row * KTOT + cb) * 2;
                union { v16bf v; uint4 q[2]; } af;
                af.q[0] = *(const uint4*)(ap);
                af.q[1] = *(const uint4*)(ap + 32);
                const char* bp = smem + WOUTF_OFF + (size_t)((kb - 2) * 32 + lane) * 32;
                union { v16bf v; uint4 q[2]; } bf_;
                bf_.q[0] = *(const uint4*)(bp);
                bf_.q[1] = *(const uint4*)(bp + 16);
                hacc = __builtin_amdgcn_wmma_f32_16x16x32_bf16(
                    false, af.v, false, bf_.v, (short)0, hacc, false, false);
            }
#pragma unroll
            for (int r = 0; r < 8; ++r) {
                int arow = mt * 16 + rbase + r;
                int ga = agbase + arow;
                float accv = hacc[r];
                float sg = sigf(accv);
                float v = (j < 2) ? accv
                        : ((j < 4) ? (0.01f + 0.2f * sg) : (0.7f * sg));
                v *= s_mask[arow];
                if (j < 5 && ga < NAG)
                    out[((size_t)t * NAG + ga) * 5 + j] = v;
            }
        }
    }
}

extern "C" void kernel_launch(void* const* d_in, const int* in_sizes, int n_in,
                              void* d_out, int out_size, void* d_ws, size_t ws_size,
                              hipStream_t stream) {
    (void)in_sizes; (void)n_in; (void)out_size; (void)d_ws; (void)ws_size;
    const float* obs    = (const float*)d_in[0];
    // d_in[1] = goals (unused by reference)
    const float* w_emb  = (const float*)d_in[2];
    const float* b_emb  = (const float*)d_in[3];
    const float* w_ih_e = (const float*)d_in[4];
    const float* b_ih_e = (const float*)d_in[5];
    const float* w_hh_e = (const float*)d_in[6];
    const float* b_hh_e = (const float*)d_in[7];
    const float* w_ih_d = (const float*)d_in[8];
    const float* b_ih_d = (const float*)d_in[9];
    const float* w_hh_d = (const float*)d_in[10];
    const float* b_hh_d = (const float*)d_in[11];
    const float* w_out  = (const float*)d_in[12];
    const float* b_out  = (const float*)d_in[13];
    float* out = (float*)d_out;

    (void)hipFuncSetAttribute((const void*)lstm_fused,
                              hipFuncAttributeMaxDynamicSharedMemorySize,
                              SMEM_TOTAL);
    dim3 grid((NAG + MB - 1) / MB);
    lstm_fused<<<grid, NTHREADS, SMEM_TOTAL, stream>>>(
        obs, w_emb, b_emb,
        w_ih_e, b_ih_e, w_hh_e, b_hh_e,
        w_ih_d, b_ih_d, w_hh_d, b_hh_d,
        w_out, b_out, out);
}
